// HPUMLAImpl_43267500540270
// MI455X (gfx1250) — matrix operand order repaired and checked by reading, not compiled, not run on hardware
//
#include <hip/hip_runtime.h>

// MI455X (gfx1250) paged-MLA attention, wave32 + v_wmma_f32_16x16x32_f16.
// Bandwidth-bound (~151 MB HBM for QK @ 23.3 TB/s ~ 6.5us; K3's V re-read
// hits the 192 MB L2). f16 matrix cores, f32 accumulation throughout.

typedef __attribute__((ext_vector_type(16))) _Float16 v16h;
typedef __attribute__((ext_vector_type(8)))  _Float16 v8h;
typedef __attribute__((ext_vector_type(4)))  _Float16 v4h;
typedef __attribute__((ext_vector_type(8)))  float    v8f;
typedef __attribute__((ext_vector_type(4)))  float    v4f;

#define B_    32
#define H_    16
#define KV_   512
#define D_    576
#define BS_   128
#define BPS_  16
#define NB_   512
#define QPAD  584            // 576 + 8 halfs -> conflict-free A-frag LDS reads
#define SCALE 0.07216878364870323f  // 192^-0.5

// ds_swizzle_b32 XOR butterfly within the 32-lane wave (group-of-32 mode:
// offset[14:10]=xor_mask, offset[4:0]=and_mask). XOR masks < 16 stay within
// each 16-lane half, matching the 16x16 C-tile row (M) grouping.
#define SWZ(x, e) \
    __int_as_float(__builtin_amdgcn_ds_swizzle(__float_as_int(x), ((e) << 10) | 0x1f))

// ---------------------------------------------------------------------------
// K1: per cache block: attn = (SCALE*q)·K^T + bias ; block max/sum ; p -> f16 ws
// grid = NB_, block = 256 (8 waves, one 16-key s-tile per wave)
// ---------------------------------------------------------------------------
__global__ __launch_bounds__(256) void k1_qk_softmax(
    const float* __restrict__ query, const float* __restrict__ key_cache,
    const float* __restrict__ block_bias, const int* __restrict__ block_list,
    const int* __restrict__ block_groups,
    float* __restrict__ ws_bm, float* __restrict__ ws_bs,
    _Float16* __restrict__ ws_p)
{
    __shared__ __align__(16) _Float16 q_lds[H_ * QPAD];
    __shared__ float wavestat[8][16];
    __shared__ float blockmax[16];

    const int n    = blockIdx.x;
    const int tid  = threadIdx.x;
    const int wave = tid >> 5;
    const int lane = tid & 31;
    const int grp  = block_groups[n];
    const int page = block_list[n];

    // stage scaled query (group's 16 heads x 576) as f16 into LDS, vectorized:
    // 16*576/4 = 2304 float4 chunks; dest v4h stores are 8B-aligned
    // (QPAD*2 = 1168 and d4*8 are both multiples of 8).
    const float* qsrc = query + (size_t)grp * H_ * D_;
    for (int i = tid; i < (H_ * D_) / 4; i += 256) {
        const int h  = i / (D_ / 4);
        const int d4 = i - h * (D_ / 4);
        v4f q4 = *(const v4f*)(qsrc + h * D_ + d4 * 4);
        v4h q4h;
        q4h[0] = (_Float16)(SCALE * q4[0]);
        q4h[1] = (_Float16)(SCALE * q4[1]);
        q4h[2] = (_Float16)(SCALE * q4[2]);
        q4h[3] = (_Float16)(SCALE * q4[3]);
        *(v4h*)(q_lds + h * QPAD + d4 * 4) = q4h;
    }
    __syncthreads();

    const float* kbase = key_cache + (size_t)page * BS_ * D_;
    const int s0    = wave * 16;        // this wave's key sub-tile
    const int lhalf = lane >> 4;        // 0: lanes 0-15, 1: lanes 16-31
    const int l15   = lane & 15;

    v8f acc = {};
    const _Float16* arow = q_lds + l15 * QPAD;   // A row M = lane&15 (head)
    const float* krow = kbase + (size_t)(s0 + l15) * D_ + lhalf * 16;
    for (int k = 0; k < 18; ++k) {               // 18 * 32 = 576 = D
        const int d0 = k * 32;
        const int o1 = d0 + lhalf * 16;
        // A fragment (16x32 f16): lane<16 -> K=d0+[0..15]; lane>=16 -> K=d0+[16..31]
        v8h alo = *(const v8h*)(arow + o1);
        v8h ahi = *(const v8h*)(arow + o1 + 8);
        v16h a;
        #pragma unroll
        for (int j = 0; j < 8; ++j) { a[j] = alo[j]; a[8 + j] = ahi[j]; }
        // B fragment (32x16): lane column N = key s0+l15, K-range = 16 contig d
        const float* kp = krow + d0;
        __builtin_prefetch(kp + 128, 0, 1);      // global_prefetch_b8, 4 steps ahead
        v16h bf;
        #pragma unroll
        for (int j = 0; j < 4; ++j) {
            v4f kq = *(const v4f*)(kp + 4 * j);
            bf[4 * j + 0] = (_Float16)kq[0];
            bf[4 * j + 1] = (_Float16)kq[1];
            bf[4 * j + 2] = (_Float16)kq[2];
            bf[4 * j + 3] = (_Float16)kq[3];
        }
        acc = __builtin_amdgcn_wmma_f32_16x16x32_f16(
                  false, a, false, bf, (short)0, acc, false, false);
    }

    // epilogue: bias (per key column), then block-wide row max / exp / row sum
    const float bias = block_bias[n * BS_ + s0 + l15];
    float x[8];
    #pragma unroll
    for (int j = 0; j < 8; ++j) x[j] = acc[j] + bias;

    #pragma unroll
    for (int j = 0; j < 8; ++j) {            // row max over the 16 key columns
        x[j] = fmaxf(x[j], SWZ(x[j], 1));
        x[j] = fmaxf(x[j], SWZ(x[j], 2));
        x[j] = fmaxf(x[j], SWZ(x[j], 4));
        x[j] = fmaxf(x[j], SWZ(x[j], 8));
    }
    if (l15 == 0) {
        #pragma unroll
        for (int j = 0; j < 8; ++j) wavestat[wave][lhalf * 8 + j] = x[j];
    }
    __syncthreads();
    if (tid < 16) {
        float m = wavestat[0][tid];
        #pragma unroll
        for (int w = 1; w < 8; ++w) m = fmaxf(m, wavestat[w][tid]);
        blockmax[tid] = m;
        ws_bm[n * H_ + tid] = m;
    }
    __syncthreads();

    float p[8];
    #pragma unroll
    for (int j = 0; j < 8; ++j) {
        const int h = j + lhalf * 8;                 // C-tile row M
        p[j] = __expf(acc[j] + bias - blockmax[h]);
        ws_p[((size_t)n * H_ + h) * BS_ + s0 + l15] = (_Float16)p[j];
    }
    float s[8];
    #pragma unroll
    for (int j = 0; j < 8; ++j) {            // row sum over the 16 key columns
        s[j] = p[j];
        s[j] += SWZ(s[j], 1);
        s[j] += SWZ(s[j], 2);
        s[j] += SWZ(s[j], 4);
        s[j] += SWZ(s[j], 8);
    }
    if (l15 == 0) {
        #pragma unroll
        for (int j = 0; j < 8; ++j) wavestat[wave][lhalf * 8 + j] = s[j];
    }
    __syncthreads();
    if (tid < 16) {
        float ssum = wavestat[0][tid];
        #pragma unroll
        for (int w = 1; w < 8; ++w) ssum += wavestat[w][tid];
        ws_bs[n * H_ + tid] = ssum;
    }
}

// ---------------------------------------------------------------------------
// K2: exact reference cross-block combine -> rescale[n][h]
// grid = 1, block = 512 (thread = (seq b, head h))
// ---------------------------------------------------------------------------
__global__ __launch_bounds__(512) void k2_combine(
    const float* __restrict__ ws_bm, const float* __restrict__ ws_bs,
    float* __restrict__ ws_rescale)
{
    const int t = threadIdx.x;
    const int b = t >> 4, h = t & 15;
    float bm[BPS_], bs[BPS_];
    #pragma unroll
    for (int j = 0; j < BPS_; ++j) {
        bm[j] = ws_bm[(b * BPS_ + j) * H_ + h];
        bs[j] = ws_bs[(b * BPS_ + j) * H_ + h];
    }
    float gm = bm[0];
    #pragma unroll
    for (int j = 1; j < BPS_; ++j) gm = fmaxf(gm, bm[j]);
    float adj[BPS_], sa[BPS_], gs = 0.f;
    #pragma unroll
    for (int j = 0; j < BPS_; ++j) {
        adj[j] = __expf(bm[j] - gm);
        sa[j]  = bs[j] * adj[j];
        gs    += sa[j];
    }
    #pragma unroll
    for (int j = 0; j < BPS_; ++j) {
        const float g = fmaxf(gs, sa[j]);   // reference's maximum() guard
        ws_rescale[(b * BPS_ + j) * H_ + h] = adj[j] / g;
    }
}

// ---------------------------------------------------------------------------
// K3: out[b] = sum_n (rescale[n]*p[n]) · V_n   (WMMA, K = 16 blocks * 128 keys)
// grid = B_*4 (seq x v-quarter), block = 256 (8 waves, one 16x16 v-tile each)
// ---------------------------------------------------------------------------
__global__ __launch_bounds__(256) void k3_pv(
    const float* __restrict__ key_cache, const int* __restrict__ block_list,
    const _Float16* __restrict__ ws_p, const float* __restrict__ ws_rescale,
    float* __restrict__ out)
{
    const int tid  = threadIdx.x;
    const int wave = tid >> 5;
    const int lane = tid & 31;
    const int b    = blockIdx.x >> 2;
    const int vq   = blockIdx.x & 3;
    const int v0   = vq * 128 + wave * 16;
    const int lhalf = lane >> 4;
    const int l15   = lane & 15;

    v8f acc = {};
    for (int j = 0; j < BPS_; ++j) {
        const int n    = b * BPS_ + j;
        const int page = block_list[n];
        const float* vbase = key_cache + (size_t)page * BS_ * D_;
        // A row M = head = lane&15 -> per-lane scalar rescale folds the combine
        const _Float16 rs = (_Float16)ws_rescale[n * H_ + l15];
        const _Float16* prow = ws_p + ((size_t)n * H_ + l15) * BS_;
        #pragma unroll
        for (int k = 0; k < 4; ++k) {            // 4 * 32 = 128 keys per block
            const int sk = k * 32;
            const int o1 = sk + lhalf * 16;
            v8h plo = *(const v8h*)(prow + o1);
            v8h phi = *(const v8h*)(prow + o1 + 8);
            v16h a;
            #pragma unroll
            for (int q = 0; q < 8; ++q) { a[q] = plo[q] * rs; a[8 + q] = phi[q] * rs; }
            // B column N = v0+l15, K-range = 16 keys (stride D_ rows)
            const int sB = sk + lhalf * 16;
            const float* vp = vbase + (size_t)sB * D_ + v0 + l15;
            __builtin_prefetch(vp + 32 * D_, 0, 1);   // next k-step's rows
            v16h bf;
            #pragma unroll
            for (int q = 0; q < 16; ++q) bf[q] = (_Float16)vp[(size_t)q * D_];
            acc = __builtin_amdgcn_wmma_f32_16x16x32_f16(
                      false, a, false, bf, (short)0, acc, false, false);
        }
    }
    float* obase = out + (size_t)b * H_ * KV_;
    #pragma unroll
    for (int j = 0; j < 8; ++j) {
        const int h = j + lhalf * 8;
        obase[(size_t)h * KV_ + v0 + l15] = acc[j];
    }
}

// ---------------------------------------------------------------------------
extern "C" void kernel_launch(void* const* d_in, const int* in_sizes, int n_in,
                              void* d_out, int out_size, void* d_ws, size_t ws_size,
                              hipStream_t stream)
{
    const float* query        = (const float*)d_in[0];
    const float* key_cache    = (const float*)d_in[1];
    // d_in[2] block_mapping: one-hot of block_groups -> redundant, unused
    const float* block_bias   = (const float*)d_in[3];
    const int*   block_list   = (const int*)d_in[4];
    const int*   block_groups = (const int*)d_in[5];
    float* out = (float*)d_out;

    // workspace layout (all fully written before read; ~2.16 MB total)
    char* ws = (char*)d_ws;
    float*    ws_bm      = (float*)(ws + 0);          // NB*H f32 = 32 KB
    float*    ws_bs      = (float*)(ws + (32 << 10)); // 32 KB
    float*    ws_rescale = (float*)(ws + (64 << 10)); // 32 KB
    _Float16* ws_p       = (_Float16*)(ws + (96 << 10)); // NB*H*BS f16 = 2 MB

    k1_qk_softmax<<<NB_, 256, 0, stream>>>(query, key_cache, block_bias,
                                           block_list, block_groups,
                                           ws_bm, ws_bs, ws_p);
    k2_combine<<<1, 512, 0, stream>>>(ws_bm, ws_bs, ws_rescale);
    k3_pv<<<B_ * 4, 256, 0, stream>>>(key_cache, block_list, ws_p, ws_rescale, out);
}
